// HetETAConvBlock_62405874811917
// MI455X (gfx1250) — matrix operand order repaired and checked by reading, compiled, not run on hardware
//
#include <hip/hip_runtime.h>
#include <hip/hip_bf16.h>
#include <hip/hip_fp16.h>

// ---------------------------------------------------------------------------
// Shapes (fixed by reference setup):
//   x      : [B=8, c0=64, T=12, N=4000] f32
//   Wconv  : [128, 64, 3, 1] f32, bconv: [128]
//   Wh     : [3, 3, 64, 64],  bh: [64]
//   Wt     : [2, 3, 64, 64],  bt: [64]
//   edges  : E=16000 per graph (src, dst, etype, w)
//   out    : [8, 128, 10, 4000] f32
// Intermediate Xn : [N=4000, BT=80, C=64] f32 (row m = n*80 + b*10 + t)
// ---------------------------------------------------------------------------

#define NN      4000
#define BB      8
#define TT      10          // T - kt + 1
#define BT      80          // BB*TT
#define CC      64
#define ROW     (BT * CC)   // 5120 floats per node row
#define MM      (NN * BT)   // 320000 GEMM rows
#define OUT_ELEMS (BB * 128 * TT * NN)   // 40,960,000

typedef __attribute__((ext_vector_type(16))) __bf16        v16bf;
typedef __attribute__((ext_vector_type(8)))  float         v8f;
typedef __attribute__((ext_vector_type(4)))  unsigned int  v4u;
typedef __attribute__((ext_vector_type(8)))  int           v8i;
typedef __attribute__((ext_vector_type(4)))  int           v4i;

// ---------------------------------------------------------------------------
// Kernel 1: temporal conv (kt=3, VALID) + residual + GLU gate, writing the
// graph-layout tensor Xn[n, b*10+t, c] directly. x-tile staged in LDS; the
// conv weights are wave-uniform (co uniform per wave32) -> scalar loads.
// ---------------------------------------------------------------------------
__global__ __launch_bounds__(256) void conv_glu_kernel(
    const float* __restrict__ x, const float* __restrict__ Wconv,
    const float* __restrict__ bconv, float* __restrict__ Xn) {
  const int ntile = blockIdx.x * 64;
  const int t = blockIdx.y;   // 0..9
  const int b = blockIdx.z;   // 0..7

  __shared__ float sX[192 * 64];   // [ci*3+dt][n-local]  48 KB

  for (int i = threadIdx.x; i < 192 * 64; i += 256) {
    const int k  = i >> 6;           // ci*3 + dt
    const int nl = i & 63;
    const int ci = k / 3, dt = k % 3;
    const int n  = ntile + nl;
    float v = 0.0f;
    if (n < NN)
      v = x[(((size_t)b * 64 + ci) * 12 + (t + dt)) * NN + n];
    sX[k * 64 + nl] = v;
  }
  __syncthreads();

  const int nl  = threadIdx.x & 63;
  const int cog = threadIdx.x >> 6;        // 0..3, wave-uniform
  const int n   = ntile + nl;
  if (n >= NN) return;

  for (int cc = 0; cc < 16; ++cc) {
    const int co = cog + cc * 4;           // 0..63, wave-uniform
    float accp = bconv[co];
    float accq = bconv[co + 64];
    const float* wp = Wconv + (size_t)co * 192;         // [co][ci][dt] contiguous
    const float* wq = Wconv + (size_t)(co + 64) * 192;
#pragma unroll 8
    for (int k = 0; k < 192; ++k) {
      const float xv = sX[k * 64 + nl];
      accp = fmaf(wp[k], xv, accp);        // scalar (uniform) weight loads
      accq = fmaf(wq[k], xv, accq);
    }
    // residual: x_in = x[b, co, t+2, n]
    const float p = accp + x[(((size_t)b * 64 + co) * 12 + (t + 2)) * NN + n];
    const float g = 1.0f / (1.0f + __expf(-accq));
    const size_t m = (size_t)n * BT + b * TT + t;
    Xn[m * CC + co] = p * g;
  }
}

// ---------------------------------------------------------------------------
// Kernel 2: initialize d_out with the biases (GEMMs then accumulate on top).
// ---------------------------------------------------------------------------
__global__ __launch_bounds__(256) void init_out_bias_kernel(
    float* __restrict__ out, const float* __restrict__ bh,
    const float* __restrict__ bt) {
  const size_t i4 = ((size_t)blockIdx.x * 256 + threadIdx.x) * 4;
  if (i4 >= OUT_ELEMS) return;
  const int ch = (int)((i4 / (TT * NN)) % 128);
  const float v = (ch < 64) ? bh[ch] : bt[ch - 64];
  *(float4*)(out + i4) = make_float4(v, v, v, v);
}

// ---------------------------------------------------------------------------
// Kernel 3: dst = alpha * src  (alpha=0 -> clear Z1, alpha=-1 -> Z2 = -Xn)
// ---------------------------------------------------------------------------
__global__ __launch_bounds__(256) void scaled_copy_kernel(
    float* __restrict__ dst, const float* __restrict__ src, float alpha) {
  const size_t i4 = ((size_t)blockIdx.x * 256 + threadIdx.x) * 4;
  if (i4 >= (size_t)MM * CC) return;
  float4 v = *(const float4*)(src + i4);
  v.x *= alpha; v.y *= alpha; v.z *= alpha; v.w *= alpha;
  *(float4*)(dst + i4) = v;
}

// ---------------------------------------------------------------------------
// Kernel 4: masked edge scatter-add (SPMM): for edges with etype == e_sel,
// Zout[dst] += scale * w[e] * Zin[src]  over the 5120-float node row.
// ---------------------------------------------------------------------------
__global__ __launch_bounds__(256) void spmm_kernel(
    const int* __restrict__ src, const int* __restrict__ dst,
    const int* __restrict__ etype, const float* __restrict__ w,
    int e_sel, float scale,
    const float* __restrict__ Zin, float* __restrict__ Zout, int E) {
  const int e = blockIdx.x;
  if (e >= E || etype[e] != e_sel) return;
  const float we = w[e] * scale;
  const float* __restrict__ in   = Zin  + (size_t)src[e] * ROW;
  float*       __restrict__ outp = Zout + (size_t)dst[e] * ROW;
  __builtin_prefetch(in + threadIdx.x * 4, 0, 0);   // global_prefetch_b8
  for (int i = threadIdx.x; i < ROW / 4; i += 256) {
    const float4 v = ((const float4*)in)[i];
    unsafeAtomicAdd(outp + i * 4 + 0, v.x * we);    // global_atomic_add_f32
    unsafeAtomicAdd(outp + i * 4 + 1, v.y * we);
    unsafeAtomicAdd(outp + i * 4 + 2, v.z * we);
    unsafeAtomicAdd(outp + i * 4 + 3, v.w * we);
  }
}

// ---------------------------------------------------------------------------
// Kernel 5: WMMA GEMM-accumulate. out(ch_base..+63) += A[320000,64] @ W[64,64]
// bf16 inputs / f32 accumulate via v_wmma_f32_16x16x32_bf16.
// The 64x64 f32 weight tile is DMA'd into LDS by the Tensor Data Mover
// (tensor_load_to_lds + s_wait_tensorcnt) issued by wave 0; fallback is a
// plain cooperative load if the builtin is unavailable.
// Block = 256 threads = 8 wave32; block owns a 128-row M-tile; each wave owns
// 16 rows x all 64 output channels (4 C-accumulators), K-loop of 2.
// Epilogue folds the back() transpose: m -> (n, b, t), out[b, ch, t, n] += c.
// ---------------------------------------------------------------------------
__global__ __launch_bounds__(256) void gemm_accum_kernel(
    const float* __restrict__ A, const float* __restrict__ W,
    float* __restrict__ out, int ch_base) {
  __shared__ float sW[64 * 64];   // 16 KB

#if defined(__HIP_DEVICE_COMPILE__) && defined(__has_builtin) && \
    __has_builtin(__builtin_amdgcn_tensor_load_to_lds)
  if (threadIdx.x < 32) {
    // Tensor DMA descriptor (ISA cdna5 ch.8): 2D tensor, 64x64 f32 tile.
    const unsigned lds_off = (unsigned)(uintptr_t)(&sW[0]);
    const unsigned long long ga = (unsigned long long)(uintptr_t)W;
    v4u g0;
    g0[0] = 1u;                                   // count=1 (valid), user mode
    g0[1] = lds_off;                              // lds_addr (bytes)
    g0[2] = (unsigned)ga;                         // global_addr[31:0]
    g0[3] = ((unsigned)(ga >> 32) & 0x01FFFFFFu)  // global_addr[56:32]
            | (2u << 30);                         // type = 2 ("image")
    v8i g1;
    g1[0] = 2 << 16;     // workgroup_mask=0, data_size=2 (4 bytes)
    g1[1] = 64 << 16;    // tensor_dim0 = 64 (lo16 in [63:48])
    g1[2] = 64 << 16;    // tensor_dim1 = 64 (lo16 in [95:80])
    g1[3] = 64 << 16;    // tile_dim0   = 64 ([127:112])
    g1[4] = 64;          // tile_dim1   = 64 ([143:128])
    g1[5] = 64;          // tensor_dim0_stride = 64 ([191:160])
    g1[6] = 0;
    g1[7] = 0;
    const v4i z4 = {0, 0, 0, 0};
#if __has_include(<hip/amd_detail/amd_gfx1250_TDM.h>)
    const v8i z8 = {0, 0, 0, 0, 0, 0, 0, 0};
    __builtin_amdgcn_tensor_load_to_lds(g0, g1, z4, z4, z8, 0);
#else
    __builtin_amdgcn_tensor_load_to_lds(g0, g1, z4, z4, 0);
#endif
    __builtin_amdgcn_s_wait_tensorcnt(0);
  }
  __syncthreads();
#else
  for (int i = threadIdx.x; i < 64 * 64; i += 256) sW[i] = W[i];
  __syncthreads();
#endif

  const int lane  = threadIdx.x & 31;
  const int wave  = threadIdx.x >> 5;
  const int mbase = blockIdx.x * 128 + wave * 16;
  const int mrow  = lane & 15;           // A-matrix row within tile
  const int khalf = (lane >> 4) * 8;     // A k-group select (ISA 16-bit A layout)

  // --- B fragments: B[k, n], n = lane&15, k = kit*32 + (lane>>4)*8 + j + (j>=8 ? 8 : 0) ---
  v16bf bfrag[2][4];
#pragma unroll
  for (int kit = 0; kit < 2; ++kit) {
#pragma unroll
    for (int ot = 0; ot < 4; ++ot) {
      const int ncol  = ot * 16 + (lane & 15);
      const int kbase = kit * 32 + (lane >> 4) * 8;
      v16bf bf;
#pragma unroll
      for (int j = 0; j < 16; ++j)
        bf[j] = (__bf16)sW[(kbase + j + (j >= 8 ? 8 : 0)) * 64 + ncol];
      bfrag[kit][ot] = bf;
    }
  }

  v8f c[4];
#pragma unroll
  for (int ot = 0; ot < 4; ++ot) c[ot] = (v8f){0.f,0.f,0.f,0.f,0.f,0.f,0.f,0.f};

  const float* __restrict__ Arow = A + (size_t)(mbase + mrow) * CC;
#pragma unroll
  for (int kit = 0; kit < 2; ++kit) {
    const int kb = kit * 32 + khalf;
    const float4 f0 = *(const float4*)(Arow + kb);
    const float4 f1 = *(const float4*)(Arow + kb + 4);
    const float4 f2 = *(const float4*)(Arow + kb + 16);
    const float4 f3 = *(const float4*)(Arow + kb + 20);
    v16bf af;
    af[0]  = (__bf16)f0.x; af[1]  = (__bf16)f0.y; af[2]  = (__bf16)f0.z; af[3]  = (__bf16)f0.w;
    af[4]  = (__bf16)f1.x; af[5]  = (__bf16)f1.y; af[6]  = (__bf16)f1.z; af[7]  = (__bf16)f1.w;
    af[8]  = (__bf16)f2.x; af[9]  = (__bf16)f2.y; af[10] = (__bf16)f2.z; af[11] = (__bf16)f2.w;
    af[12] = (__bf16)f3.x; af[13] = (__bf16)f3.y; af[14] = (__bf16)f3.z; af[15] = (__bf16)f3.w;
#pragma unroll
    for (int ot = 0; ot < 4; ++ot)
      c[ot] = __builtin_amdgcn_wmma_f32_16x16x32_bf16(
          false, af, false, bfrag[kit][ot], (short)0, c[ot], false, false);
  }

  // Epilogue. C layout: reg r holds (M = r + (lane>>4)*8, N = lane&15).
  // r-outer / ot-inner so the m -> (node,b,t) divisions run once per row and
  // the 4 o-tiles become constant-stride offsets (ot * 16 * TT * NN).
  const int mo   = (lane >> 4) * 8;
  const int ncol = lane & 15;
#pragma unroll
  for (int r = 0; r < 8; ++r) {
    const unsigned m    = (unsigned)(mbase + mo + r);
    const unsigned node = m / BT;
    const unsigned rr   = m % BT;
    const unsigned bb   = rr / TT;
    const unsigned tt   = rr % TT;
    float* obase = out + (((size_t)bb * 128 + ch_base + ncol) * TT + tt) * NN + node;
#pragma unroll
    for (int ot = 0; ot < 4; ++ot)
      obase[(size_t)ot * 16 * TT * NN] += c[ot][r];
  }
}

// ---------------------------------------------------------------------------
// Host orchestration
// ---------------------------------------------------------------------------
extern "C" void kernel_launch(void* const* d_in, const int* in_sizes, int n_in,
                              void* d_out, int out_size, void* d_ws, size_t ws_size,
                              hipStream_t stream) {
  const float* x     = (const float*)d_in[0];
  const float* Wconv = (const float*)d_in[1];
  const float* bconv = (const float*)d_in[2];
  const float* Wh    = (const float*)d_in[3];
  const float* bh    = (const float*)d_in[4];
  const float* Wt    = (const float*)d_in[5];
  const float* bt    = (const float*)d_in[6];
  const int*   g_src[2] = { (const int*)d_in[7],  (const int*)d_in[11] };
  const int*   g_dst[2] = { (const int*)d_in[8],  (const int*)d_in[12] };
  const int*   g_ety[2] = { (const int*)d_in[9],  (const int*)d_in[13] };
  const float* g_w[2]   = { (const float*)d_in[10], (const float*)d_in[14] };
  const float* g_W[2]   = { Wh, Wt };
  const int    g_net[2] = { in_sizes[3] / (3 * 64 * 64), in_sizes[5] / (3 * 64 * 64) };
  const int    g_chb[2] = { 0, 64 };
  const int    E        = in_sizes[7];

  float* out = (float*)d_out;
  float* Xn  = (float*)d_ws;                       // [320000, 64] f32
  float* Z1  = Xn + (size_t)MM * CC;               // spmm level-1
  float* Z2  = Z1 + (size_t)MM * CC;               // spmm level-2

  const int copyBlocks = (MM * CC / 4 + 255) / 256;        // 20000
  const int outBlocks  = (OUT_ELEMS / 4 + 255) / 256;      // 40000
  const int gemmBlocks = MM / 128;                          // 2500

  // 1. temporal conv + GLU -> Xn
  conv_glu_kernel<<<dim3((NN + 63) / 64, TT, BB), 256, 0, stream>>>(x, Wconv, bconv, Xn);

  // 2. out = bias
  init_out_bias_kernel<<<outBlocks, 256, 0, stream>>>(out, bh, bt);

  // 3. per-graph, per-etype Chebyshev recursion + WMMA projections
  for (int g = 0; g < 2; ++g) {
    for (int e = 0; e < g_net[g]; ++e) {
      const float* We = g_W[g] + (size_t)e * 3 * 64 * 64;
      // k = 0 : out += Xn @ W[e,0]
      gemm_accum_kernel<<<gemmBlocks, 256, 0, stream>>>(Xn, We, out, g_chb[g]);
      // Z1 = A_e Xn
      scaled_copy_kernel<<<copyBlocks, 256, 0, stream>>>(Z1, Xn, 0.0f);
      spmm_kernel<<<E, 256, 0, stream>>>(g_src[g], g_dst[g], g_ety[g], g_w[g],
                                         e, 1.0f, Xn, Z1, E);
      // k = 1 : out += Z1 @ W[e,1]
      gemm_accum_kernel<<<gemmBlocks, 256, 0, stream>>>(Z1, We + 64 * 64, out, g_chb[g]);
      // Z2 = 2 A_e Z1 - Xn
      scaled_copy_kernel<<<copyBlocks, 256, 0, stream>>>(Z2, Xn, -1.0f);
      spmm_kernel<<<E, 256, 0, stream>>>(g_src[g], g_dst[g], g_ety[g], g_w[g],
                                         e, 2.0f, Z1, Z2, E);
      // k = 2 : out += Z2 @ W[e,2]
      gemm_accum_kernel<<<gemmBlocks, 256, 0, stream>>>(Z2, We + 2 * 64 * 64, out, g_chb[g]);
    }
  }
}
